// VoxelAttention_55748675502260
// MI455X (gfx1250) — matrix operand order repaired and checked by reading, not compiled, not run on hardware
//
#include <hip/hip_runtime.h>
#include <hip/hip_bf16.h>

typedef __attribute__((ext_vector_type(16))) __bf16 v16bf;
typedef __attribute__((ext_vector_type(8)))  float  v8f;
typedef unsigned short u16;
typedef unsigned int   u32;

// Problem constants
constexpr int Bc  = 2;
constexpr int NVc = 8192;
constexpr int Cc  = 256;
constexpr int NPc = 2048;
constexpr int NSc = 32;
constexpr int Hc  = 4;
constexpr int DHc = 64;

union V16U { v16bf v; u16 u[16]; u32 w[8]; uint4 q[2]; };
union Pk8  { uint4 q; u16 u[8]; u32 w[4]; };

static __device__ __forceinline__ u16 f2bf(float f) {
  u32 u = __float_as_uint(f);
  u32 r = (u + 0x7FFFu + ((u >> 16) & 1u)) >> 16;   // RNE (used only in prep)
  return (u16)r;
}
// Packed 2xfloat -> 2xbf16 in one dword: +0x8000 round bias, then one
// v_perm_b32 grabs both high halves. 3 VALU per pair, no b16 moves.
static __device__ __forceinline__ u32 pkbf(float a, float b) {
  u32 ua = __float_as_uint(a) + 0x8000u;
  u32 ub = __float_as_uint(b) + 0x8000u;
  return __builtin_amdgcn_perm(ub, ua, 0x07060302u);
}
static __device__ __forceinline__ float bf2f(u32 x) {
  return __uint_as_float(x << 16);
}
static __device__ __forceinline__ void pack8w(u32* dst, float4 a, float4 b) {
  dst[0] = pkbf(a.x, a.y); dst[1] = pkbf(a.z, a.w);
  dst[2] = pkbf(b.x, b.y); dst[3] = pkbf(b.z, b.w);
}
static __device__ __forceinline__ float4 add4(float4 a, float4 b) {
  return make_float4(a.x+b.x, a.y+b.y, a.z+b.z, a.w+b.w);
}
static __device__ __forceinline__ v8f wmma_bf16(const V16U& a, const V16U& b, v8f c) {
  return __builtin_amdgcn_wmma_f32_16x16x32_bf16(false, a.v, false, b.v, (short)0, c, false, false);
}

// ---------------------------------------------------------------------------
// Kernel 1: weight prep. Wk,Wv kept (n,k) row-major (A-operand).
//           Wq,Wo also kept (n,k) row-major (B-operand). All bf16.
// ---------------------------------------------------------------------------
__global__ void prep_weights(const float* __restrict__ Wq, const float* __restrict__ Wk,
                             const float* __restrict__ Wv, const float* __restrict__ Wo,
                             u16* __restrict__ wk_bf, u16* __restrict__ wv_bf,
                             u16* __restrict__ wqt,   u16* __restrict__ wot) {
  int t = blockIdx.x * blockDim.x + threadIdx.x;     // 65536 total
  int i = t >> 8, j = t & 255;                       // row i (fan_out), col j (fan_in)
  wk_bf[t] = f2bf(Wk[t]);
  wv_bf[t] = f2bf(Wv[t]);
  wqt[t] = f2bf(Wq[t]);
  wot[t] = f2bf(Wo[t]);
}

// ---------------------------------------------------------------------------
// Kernel 2: Q projection. A = q_feat rows (fp32 -> bf16), B = wqt, stored (n,k).
// Output fp32 (B,H,NV,DH). One wave: 16 rows x 64 cols.
// ---------------------------------------------------------------------------
__global__ void qproj(const float* __restrict__ qf, const u16* __restrict__ wqt,
                      const float* __restrict__ bq, float* __restrict__ Qout) {
  int wid  = blockIdx.x * 8 + (threadIdx.x >> 5);    // 4096 waves
  int lane = threadIdx.x & 31;
  int ng   = wid & 3;
  int tm   = wid >> 2;
  int row0 = tm * 16;                                // global row in [0, B*NV)
  int n0   = ng * 64;
  int mrow = row0 + (lane & 15);
  int asel = (lane >> 4) * 8;

  v8f acc[4];
  v8f zero = {0.f,0.f,0.f,0.f,0.f,0.f,0.f,0.f};
#pragma unroll
  for (int t = 0; t < 4; ++t) acc[t] = zero;

  for (int kk = 0; kk < 8; ++kk) {
    int kb = kk * 32;
    int krow = kb + lane;
    // Issue all B loads first (clause + staggered waits).
    V16U bm[4];
#pragma unroll
    for (int t = 0; t < 4; ++t) {
      const u16* bp = wqt + (size_t)(n0 + t * 16 + (lane & 15)) * 256 + kb + asel;
      bm[t].q[0] = *(const uint4*)bp;
      bm[t].q[1] = *(const uint4*)(bp + 16);
    }
    // A operand (fp32 -> bf16 packed convert).
    const float* ap = qf + (size_t)mrow * 256 + kb + asel;
    float4 f0 = *(const float4*)(ap);
    float4 f1 = *(const float4*)(ap + 4);
    float4 f2 = *(const float4*)(ap + 16);
    float4 f3 = *(const float4*)(ap + 20);
    V16U a;
    pack8w(a.w,     f0, f1);
    pack8w(a.w + 4, f2, f3);
#pragma unroll
    for (int t = 0; t < 4; ++t) acc[t] = wmma_bf16(a, bm[t], acc[t]);
  }
#pragma unroll
  for (int t = 0; t < 4; ++t) {
    int n = n0 + t * 16 + (lane & 15);
    float bias = bq[n];
    int h = n >> 6, d = n & 63;
#pragma unroll
    for (int r = 0; r < 8; ++r) {
      int rowg = row0 + r + 8 * (lane >> 4);
      int b  = rowg >> 13;
      int nv = rowg & (NVc - 1);
      Qout[(((size_t)b * Hc + h) * NVc + nv) * DHc + d] = acc[t][r] + bias;
    }
  }
}

// ---------------------------------------------------------------------------
// Kernel 3: K/V projection, transposed GEMM: O^T[n][ps] = W[n][:] . slab[:][ps]
// A = weight rows staged in LDS (all 8 waves of a block share the same 64
// output channels); B = (gf+kp)/gf columns (fp32 -> bf16), contiguous loads.
// One wave: 64 n-values x 16 ps, for BOTH K and V (shares gf/kp loads).
// Stores bf16 (B,H,NP,NS,DH), one aligned 16B vector per lane per tile.
// ---------------------------------------------------------------------------
__global__ void kvproj(const float* __restrict__ gf, const float* __restrict__ kpos,
                       const u16* __restrict__ wkb, const u16* __restrict__ wvb,
                       const float* __restrict__ bk, const float* __restrict__ bv,
                       u16* __restrict__ Kout, u16* __restrict__ Vout) {
  __shared__ u16 lwk[64 * 256];                      // 32 KB
  __shared__ u16 lwv[64 * 256];                      // 32 KB
  int wslot = threadIdx.x >> 5;
  int lane  = threadIdx.x & 31;
  int wid = blockIdx.x * 8 + wslot;                  // 32768 waves
  int pst = wid & 4095;
  int g   = (wid >> 12) & 3;                         // uniform per block
  int b   = wid >> 14;                               // uniform per block
  int ps0 = pst * 16;
  int n0  = g * 64;
  const size_t slab = (size_t)Cc * NPc * NSc;
  int asel = (lane >> 4) * 8;

  // Cooperative LDS fill: 2048 uint4 per matrix, 8 per thread.
  {
    const uint4* sk = (const uint4*)(wkb + (size_t)n0 * 256);
    const uint4* sv = (const uint4*)(wvb + (size_t)n0 * 256);
    uint4* dk = (uint4*)lwk;
    uint4* dv = (uint4*)lwv;
#pragma unroll
    for (int i = 0; i < 8; ++i) {
      int e = threadIdx.x + i * 256;
      dk[e] = sk[e];
      dv[e] = sv[e];
    }
  }
  __syncthreads();

  v8f accK[4], accV[4];
  v8f zero = {0.f,0.f,0.f,0.f,0.f,0.f,0.f,0.f};
#pragma unroll
  for (int t = 0; t < 4; ++t) { accK[t] = zero; accV[t] = zero; }

  for (int kk = 0; kk < 8; ++kk) {
    int kb = kk * 32;
    // B-operand: lane holds ps = ps0 + lane%16, k = kb+asel+{0..7,16..23}.
    const size_t S = (size_t)NPc * NSc;
    const float* gp = gf   + (size_t)b * slab + (size_t)(kb + asel) * S + ps0 + (lane & 15);
    const float* pp = kpos + (size_t)b * slab + (size_t)(kb + asel) * S + ps0 + (lane & 15);
    float4 g0 = make_float4(gp[0], gp[S], gp[2*S], gp[3*S]);
    float4 g1 = make_float4(gp[4*S], gp[5*S], gp[6*S], gp[7*S]);
    float4 g2 = make_float4(gp[16*S], gp[17*S], gp[18*S], gp[19*S]);
    float4 g3 = make_float4(gp[20*S], gp[21*S], gp[22*S], gp[23*S]);
    float4 p0 = make_float4(pp[0], pp[S], pp[2*S], pp[3*S]);
    float4 p1 = make_float4(pp[4*S], pp[5*S], pp[6*S], pp[7*S]);
    float4 p2 = make_float4(pp[16*S], pp[17*S], pp[18*S], pp[19*S]);
    float4 p3 = make_float4(pp[20*S], pp[21*S], pp[22*S], pp[23*S]);
    // A-operands from LDS (ds_load_b128).
    V16U aK[4], aV[4];
#pragma unroll
    for (int t = 0; t < 4; ++t) {
      int off = (t * 16 + (lane & 15)) * 256 + kb + asel;
      aK[t].q[0] = *(const uint4*)(lwk + off);
      aK[t].q[1] = *(const uint4*)(lwk + off + 16);
      aV[t].q[0] = *(const uint4*)(lwv + off);
      aV[t].q[1] = *(const uint4*)(lwv + off + 16);
    }
    V16U bV, bK;
    pack8w(bV.w,     g0, g1);
    pack8w(bV.w + 4, g2, g3);
    pack8w(bK.w,     add4(g0,p0), add4(g1,p1));
    pack8w(bK.w + 4, add4(g2,p2), add4(g3,p3));
#pragma unroll
    for (int t = 0; t < 4; ++t) {
      accK[t] = wmma_bf16(aK[t], bK, accK[t]);
      accV[t] = wmma_bf16(aV[t], bV, accV[t]);
    }
  }
  // Epilogue: bias + bf16 pack, one aligned 16B vector store per lane per tile.
  int psl = ps0 + (lane & 15);
  int p = psl >> 5, s = psl & (NSc - 1);
#pragma unroll
  for (int t = 0; t < 4; ++t) {
    int d0 = t * 16 + 8 * (lane >> 4);               // h == g for the whole wave
    Pk8 pk, pv;
#pragma unroll
    for (int i = 0; i < 4; ++i) {
      int n = n0 + d0 + 2 * i;
      pk.w[i] = pkbf(accK[t][2*i] + bk[n], accK[t][2*i+1] + bk[n+1]);
      pv.w[i] = pkbf(accV[t][2*i] + bv[n], accV[t][2*i+1] + bv[n+1]);
    }
    size_t base = ((((size_t)b * Hc + g) * NPc + p) * NSc + s) * DHc + d0;
    *(uint4*)(Kout + base) = pk.q;
    *(uint4*)(Vout + base) = pv.q;
  }
}

// ---------------------------------------------------------------------------
// Kernel 4: gathered attention. One wave per (b,h,n). lane = s for scores,
// wave-wide shfl softmax over NS=32, lane = d-pair for the V reduction.
// K/V (134 MB bf16 total) are L2-resident on MI455X (192 MB L2).
// ---------------------------------------------------------------------------
__global__ void attn_kernel(const float* __restrict__ Q, const u16* __restrict__ K,
                            const u16* __restrict__ V, const int* __restrict__ idx,
                            float* __restrict__ AO) {
  __shared__ float qsh[8][64];
  __shared__ float ash[8][32];
  int wslot = threadIdx.x >> 5;
  int lane  = threadIdx.x & 31;
  int wid = blockIdx.x * 8 + wslot;                  // 65536 waves
  int n = wid & (NVc - 1);
  int h = (wid >> 13) & (Hc - 1);
  int b = wid >> 15;

  const float* qp = Q + (((size_t)b * Hc + h) * NVc + n) * DHc;
  float2 q2 = *(const float2*)(qp + 2 * lane);
  qsh[wslot][2 * lane]     = q2.x;
  qsh[wslot][2 * lane + 1] = q2.y;
  __syncthreads();

  int p = idx[((size_t)b * Hc + h) * NVc + n];
  const u16* kp_ = K + ((((size_t)b * Hc + h) * NPc + p) * NSc + lane) * DHc;
  float acc = 0.f;
#pragma unroll
  for (int i = 0; i < 8; ++i) {
    Pk8 kq; kq.q = *(const uint4*)(kp_ + i * 8);
#pragma unroll
    for (int j = 0; j < 8; ++j) acc += qsh[wslot][i * 8 + j] * bf2f(kq.u[j]);
  }
  float sc = acc * 0.125f;                           // 1/sqrt(64)
  float m = sc;
#pragma unroll
  for (int o = 16; o >= 1; o >>= 1) m = fmaxf(m, __shfl_xor(m, o, 32));
  float e = __expf(sc - m);
  float ssum = e;
#pragma unroll
  for (int o = 16; o >= 1; o >>= 1) ssum += __shfl_xor(ssum, o, 32);
  ash[wslot][lane] = e / ssum;
  __syncthreads();

  const u16* vp = V + ((((size_t)b * Hc + h) * NPc + p) * NSc) * DHc + 2 * lane;
  float o0 = 0.f, o1 = 0.f;
#pragma unroll
  for (int s = 0; s < NSc; ++s) {
    float a = ash[wslot][s];
    u32 u = *(const u32*)(vp + (size_t)s * DHc);
    o0 += a * bf2f(u & 0xffffu);
    o1 += a * bf2f(u >> 16);
  }
  float* op = AO + ((size_t)b * NVc + n) * Cc + h * DHc + 2 * lane;
  *(float2*)op = make_float2(o0, o1);
}

// ---------------------------------------------------------------------------
// Kernel 5: output projection. A = AO rows (fp32 -> bf16), B = wot, stored (n,k).
// ---------------------------------------------------------------------------
__global__ void oproj(const float* __restrict__ AO, const u16* __restrict__ wot,
                      const float* __restrict__ bo, float* __restrict__ out) {
  int wid  = blockIdx.x * 8 + (threadIdx.x >> 5);    // 4096 waves
  int lane = threadIdx.x & 31;
  int ng   = wid & 3;
  int tm   = wid >> 2;
  int row0 = tm * 16;
  int n0   = ng * 64;
  int mrow = row0 + (lane & 15);
  int asel = (lane >> 4) * 8;

  v8f acc[4];
  v8f zero = {0.f,0.f,0.f,0.f,0.f,0.f,0.f,0.f};
#pragma unroll
  for (int t = 0; t < 4; ++t) acc[t] = zero;

  for (int kk = 0; kk < 8; ++kk) {
    int kb = kk * 32;
    int krow = kb + lane;
    V16U bm[4];
#pragma unroll
    for (int t = 0; t < 4; ++t) {
      const u16* bp = wot + (size_t)(n0 + t * 16 + (lane & 15)) * 256 + kb + asel;
      bm[t].q[0] = *(const uint4*)bp;
      bm[t].q[1] = *(const uint4*)(bp + 16);
    }
    const float* ap = AO + (size_t)mrow * 256 + kb + asel;
    float4 f0 = *(const float4*)(ap);
    float4 f1 = *(const float4*)(ap + 4);
    float4 f2 = *(const float4*)(ap + 16);
    float4 f3 = *(const float4*)(ap + 20);
    V16U a;
    pack8w(a.w,     f0, f1);
    pack8w(a.w + 4, f2, f3);
#pragma unroll
    for (int t = 0; t < 4; ++t) acc[t] = wmma_bf16(a, bm[t], acc[t]);
  }
#pragma unroll
  for (int t = 0; t < 4; ++t) {
    int nn = n0 + t * 16 + (lane & 15);
    float bias = bo[nn];
#pragma unroll
    for (int r = 0; r < 8; ++r) {
      int rowg = row0 + r + 8 * (lane >> 4);
      out[(size_t)rowg * 256 + nn] = acc[t][r] + bias;
    }
  }
}

// ---------------------------------------------------------------------------
extern "C" void kernel_launch(void* const* d_in, const int* in_sizes, int n_in,
                              void* d_out, int out_size, void* d_ws, size_t ws_size,
                              hipStream_t stream) {
  const float* q_feat     = (const float*)d_in[0];
  const float* group_feat = (const float*)d_in[1];
  const float* k_pos      = (const float*)d_in[2];
  const int*   nidx       = (const int*)d_in[3];
  const float* Wq = (const float*)d_in[4];  const float* bq = (const float*)d_in[5];
  const float* Wk = (const float*)d_in[6];  const float* bk = (const float*)d_in[7];
  const float* Wv = (const float*)d_in[8];  const float* bv = (const float*)d_in[9];
  const float* Wo = (const float*)d_in[10]; const float* bo = (const float*)d_in[11];

  char* ws = (char*)d_ws;
  u16*   wk_bf = (u16*)(ws + 0);
  u16*   wv_bf = (u16*)(ws + 131072);
  u16*   wqt   = (u16*)(ws + 262144);
  u16*   wot   = (u16*)(ws + 393216);
  float* Qws   = (float*)(ws + 524288);                         // 16 MB
  u16*   Kws   = (u16*)(ws + 524288 + 16777216);                // 64 MB
  u16*   Vws   = (u16*)(ws + 524288 + 16777216 + 67108864);     // 64 MB
  float* AOws  = (float*)(ws + 524288 + 16777216 + 2*67108864); // 16 MB

  prep_weights<<<256, 256, 0, stream>>>(Wq, Wk, Wv, Wo, wk_bf, wv_bf, wqt, wot);
  qproj<<<512, 256, 0, stream>>>(q_feat, wqt, bq, Qws);
  kvproj<<<4096, 256, 0, stream>>>(group_feat, k_pos, wk_bf, wv_bf, bk, bv, Kws, Vws);
  attn_kernel<<<8192, 256, 0, stream>>>(Qws, Kws, Vws, nidx, AOws);
  oproj<<<512, 256, 0, stream>>>(AOws, wot, bo, (float*)d_out);
}
